// MultiHeadAttentionBlock_44143673868715
// MI455X (gfx1250) — matrix operand order repaired and checked by reading, not compiled, not run on hardware
//
#include <hip/hip_runtime.h>
#include <math.h>

typedef _Float16 half_t;
typedef __attribute__((ext_vector_type(16))) _Float16 v16h;
typedef __attribute__((ext_vector_type(8)))  _Float16 v8h;
typedef __attribute__((ext_vector_type(8)))  float    v8f;

constexpr float EPS_RMS = 1e-6f;
constexpr float EPS_LN  = 1e-5f;
constexpr float SOFTCAP_F = 5.0f;
constexpr float ROPE_BASE_F = 8192.0f;

constexpr int BB = 2;
constexpr int SS = 2048;
constexpr int DD = 1536;
constexpr int HH = 8;
constexpr int DQn = 128;
constexpr int DKn = 128;
constexpr int DVn = 192;
constexpr int NQKV = 1024 + 128 + 192;   // 1344
constexpr int MM = BB * SS;              // 4096

// ---------------------------------------------------------------------------
// Elementwise: RMS-BN (eval) + cast to f16
// ---------------------------------------------------------------------------
__global__ void __launch_bounds__(256)
prenorm_cast(const float* __restrict__ x, const float* __restrict__ g,
             const float* __restrict__ b, const float* __restrict__ rr,
             half_t* __restrict__ out, int n, int D) {
  int i = blockIdx.x * blockDim.x + threadIdx.x;
  if (i >= n) return;
  int c = i % D;
  float v = g[c] * (x[i] * rsqrtf(rr[c] + EPS_RMS)) + b[c];
  out[i] = (half_t)v;
}

// Pack Wq|Wk|Wv -> f16 [D x 1344]
__global__ void __launch_bounds__(256)
pack_qkv(const float* __restrict__ Wq, const float* __restrict__ Wk,
         const float* __restrict__ Wv, half_t* __restrict__ out, int n) {
  int i = blockIdx.x * blockDim.x + threadIdx.x;
  if (i >= n) return;
  int k = i / NQKV, c = i % NQKV;
  float v;
  if (c < 1024)      v = Wq[k * 1024 + c];
  else if (c < 1152) v = Wk[k * 128 + (c - 1024)];
  else               v = Wv[k * 192 + (c - 1152)];
  out[i] = (half_t)v;
}

__global__ void __launch_bounds__(256)
cast_f32_f16(const float* __restrict__ a, half_t* __restrict__ o, int n) {
  int i = blockIdx.x * blockDim.x + threadIdx.x;
  if (i < n) o[i] = (half_t)a[i];
}

// ---------------------------------------------------------------------------
// f16 WMMA GEMM: C[M,N] = A[M,K] * B[K,N] (+ optional fused bias+RMS epilogue)
// BM=64, BN=64, BK=32, 128 threads = 4 waves, wave -> 16 rows x 64 cols.
// A tile staged via gfx1250 async direct-to-LDS DMA (ASYNCcnt path);
// B tile transposed in flight through VGPRs.
// ---------------------------------------------------------------------------
__global__ void __launch_bounds__(128)
gemm_f16(const half_t* __restrict__ A, const half_t* __restrict__ Bm,
         float* __restrict__ C, int M, int N, int K,
         const float* __restrict__ bias, const float* __restrict__ gamma,
         const float* __restrict__ beta, const float* __restrict__ rr) {
  __shared__ half_t lA[64][40];    // [row][k], padded pitch
  __shared__ half_t lBt[64][40];   // [n][k] (B staged transposed)
  const int t = threadIdx.x;
  const int w = t >> 5;
  const int lane = t & 31;
  const int ln = lane & 15;
  const int hf = lane >> 4;
  const int row0 = blockIdx.x * 64;
  const int n0 = blockIdx.y * 64;

  v8f zero = {0.f, 0.f, 0.f, 0.f, 0.f, 0.f, 0.f, 0.f};
  v8f acc[4];
  for (int i = 0; i < 4; ++i) acc[i] = zero;

  const int nk = K >> 5;
  for (int kb = 0; kb < nk; ++kb) {
    const int k0 = kb << 5;
    // stage A tile 64x32: async global->LDS b128 (no VGPR round-trip)
    for (int i = 0; i < 2; ++i) {
      int u = t + i * 128;
      int r = u >> 2, cg = (u & 3) << 3;
      unsigned ldsa = (unsigned)(uintptr_t)&lA[r][cg];
      unsigned long long ga =
          (unsigned long long)(uintptr_t)(A + (size_t)(row0 + r) * K + k0 + cg);
      asm volatile("global_load_async_to_lds_b128 %0, %1, off"
                   :: "v"(ldsa), "v"(ga) : "memory");
    }
    // stage B tile 32x64 transposed (through VGPRs)
    for (int i = 0; i < 2; ++i) {
      int u = t + i * 128;
      int kr = u >> 3, ng = (u & 7) << 3;
      v8h val = *(const v8h*)(Bm + (size_t)(k0 + kr) * N + n0 + ng);
      for (int j = 0; j < 8; ++j) lBt[ng + j][kr] = val[j];
    }
    if (kb + 1 < nk) {   // global_prefetch_b8, near-cache scope
      __builtin_prefetch(A + (size_t)(row0 + (t >> 1)) * K + k0 + 32, 0, 3);
      __builtin_prefetch(Bm + (size_t)(k0 + 32 + (t & 31)) * N + n0, 0, 3);
    }
    asm volatile("s_wait_asynccnt 0x0" ::: "memory");
    __syncthreads();

    // A fragment: lane row = w*16+ln; elem e -> k = (e%8)+16*(e/8)+8*hf
    v16h af;
    {
      int r = (w << 4) + ln;
      v8h c0 = *(const v8h*)&lA[r][hf << 3];
      v8h c1 = *(const v8h*)&lA[r][16 + (hf << 3)];
      for (int e = 0; e < 8; ++e) { af[e] = c0[e]; af[e + 8] = c1[e]; }
    }
    for (int ns = 0; ns < 4; ++ns) {
      // B fragment: lane col = ns*16+ln; elem e -> k = (e%8)+16*(e/8)+8*hf
      int n = (ns << 4) + ln;
      v16h bf;
      v8h c0 = *(const v8h*)&lBt[n][hf << 3];
      v8h c1 = *(const v8h*)&lBt[n][16 + (hf << 3)];
      for (int e = 0; e < 8; ++e) { bf[e] = c0[e]; bf[e + 8] = c1[e]; }
      acc[ns] = __builtin_amdgcn_wmma_f32_16x16x32_f16(
          false, af, false, bf, (short)0, acc[ns], false, false);
    }
    __syncthreads();
  }

  // store (optionally fused: out = g2*((acc+bo)*rsqrt(rr2+eps)) + b2)
  for (int ns = 0; ns < 4; ++ns) {
    int col = n0 + (ns << 4) + ln;
    for (int r = 0; r < 8; ++r) {
      int rowg = row0 + (w << 4) + r + (hf << 3);
      float v = acc[ns][r];
      if (gamma) v = gamma[col] * ((v + bias[col]) * rsqrtf(rr[col] + EPS_RMS)) + beta[col];
      C[(size_t)rowg * N + col] = v;
    }
  }
}

// ---------------------------------------------------------------------------
// Per-vector LayerNorm (+RoPE) post-processing. One wave per vector.
// ---------------------------------------------------------------------------
__global__ void __launch_bounds__(128)
q_post(const float* __restrict__ qkv, const float* __restrict__ qg,
       const float* __restrict__ qb, half_t* __restrict__ qh) {
  int vid = blockIdx.x * 4 + (threadIdx.x >> 5);   // vid = (b*H+h)*S + s
  int lane = threadIdx.x & 31;
  int s = vid % SS;
  int bh = vid / SS;
  int h = bh % HH, b = bh / HH;
  const float* src = qkv + (size_t)(b * SS + s) * NQKV + h * DQn;
  float v[4], sum = 0.f, sq = 0.f;
  for (int i = 0; i < 4; ++i) { v[i] = src[lane + 32 * i]; sum += v[i]; sq += v[i] * v[i]; }
  for (int m = 16; m > 0; m >>= 1) { sum += __shfl_xor(sum, m, 32); sq += __shfl_xor(sq, m, 32); }
  float mu = sum * (1.f / 128.f);
  float inv = rsqrtf(sq * (1.f / 128.f) - mu * mu + EPS_LN);
  for (int i = 0; i < 4; ++i)
    v[i] = qg[lane + 32 * i] * ((v[i] - mu) * inv) + qb[lane + 32 * i];
  float out[4];
  for (int p = 0; p < 2; ++p) {   // pairs (d, d+64) are local to the lane
    int j = lane + 32 * p;
    float fr = expf(-logf(ROPE_BASE_F) * (float)j * (1.f / 64.f));
    float a = (float)s * fr;
    float c = cosf(a), sn = sinf(a);
    out[p]     = v[p] * c - v[p + 2] * sn;
    out[p + 2] = v[p] * sn + v[p + 2] * c;
  }
  half_t* dst = qh + (size_t)vid * DQn;
  for (int i = 0; i < 4; ++i) dst[lane + 32 * i] = (half_t)out[i];
}

__global__ void __launch_bounds__(128)
k_post(const float* __restrict__ qkv, const float* __restrict__ kg,
       const float* __restrict__ kb2, half_t* __restrict__ kh) {
  int vid = blockIdx.x * 4 + (threadIdx.x >> 5);   // vid = b*S + s
  int lane = threadIdx.x & 31;
  int s = vid % SS;
  const float* src = qkv + (size_t)vid * NQKV + 1024;
  float v[4], sum = 0.f, sq = 0.f;
  for (int i = 0; i < 4; ++i) { v[i] = src[lane + 32 * i]; sum += v[i]; sq += v[i] * v[i]; }
  for (int m = 16; m > 0; m >>= 1) { sum += __shfl_xor(sum, m, 32); sq += __shfl_xor(sq, m, 32); }
  float mu = sum * (1.f / 128.f);
  float inv = rsqrtf(sq * (1.f / 128.f) - mu * mu + EPS_LN);
  for (int i = 0; i < 4; ++i)
    v[i] = kg[lane + 32 * i] * ((v[i] - mu) * inv) + kb2[lane + 32 * i];
  float out[4];
  for (int p = 0; p < 2; ++p) {
    int j = lane + 32 * p;
    float fr = expf(-logf(ROPE_BASE_F) * (float)j * (1.f / 64.f));
    float a = (float)s * fr;
    float c = cosf(a), sn = sinf(a);
    out[p]     = v[p] * c - v[p + 2] * sn;
    out[p + 2] = v[p] * sn + v[p + 2] * c;
  }
  half_t* dst = kh + (size_t)vid * DKn;
  for (int i = 0; i < 4; ++i) dst[lane + 32 * i] = (half_t)out[i];
}

__global__ void __launch_bounds__(128)
v_post(const float* __restrict__ qkv, const float* __restrict__ vg,
       const float* __restrict__ vb2, half_t* __restrict__ vh) {
  int vid = blockIdx.x * 4 + (threadIdx.x >> 5);   // vid = b*S + s
  int lane = threadIdx.x & 31;
  const float* src = qkv + (size_t)vid * NQKV + 1152;
  float v[6], sum = 0.f, sq = 0.f;
  for (int i = 0; i < 6; ++i) { v[i] = src[lane + 32 * i]; sum += v[i]; sq += v[i] * v[i]; }
  for (int m = 16; m > 0; m >>= 1) { sum += __shfl_xor(sum, m, 32); sq += __shfl_xor(sq, m, 32); }
  float mu = sum * (1.f / 192.f);
  float inv = rsqrtf(sq * (1.f / 192.f) - mu * mu + EPS_LN);
  half_t* dst = vh + (size_t)vid * DVn;
  for (int i = 0; i < 6; ++i)
    dst[lane + 32 * i] = (half_t)(vg[lane + 32 * i] * ((v[i] - mu) * inv) + vb2[lane + 32 * i]);
}

// ---------------------------------------------------------------------------
// Flash attention: grid (S/64, B*H), 128 thr = 4 waves, wave = 16 query rows.
// Streams keys in 32-wide tiles; online softmax with tanh softcap + bias.
// ---------------------------------------------------------------------------
__global__ void __launch_bounds__(128)
attention(const half_t* __restrict__ qh, const half_t* __restrict__ kh,
          const half_t* __restrict__ vh, const float* __restrict__ bias,
          half_t* __restrict__ yh) {
  __shared__ half_t vtl[192][40];     // V tile transposed: [c][k]
  __shared__ half_t pbuf[4][16][40];  // per-wave P 16x32
  const int t = threadIdx.x, w = t >> 5, lane = t & 31;
  const int ln = lane & 15, hf = lane >> 4;
  const int s0 = blockIdx.x * 64;
  const int bh = blockIdx.y;
  const int b = bh / HH, h = bh % HH;
  const half_t* qbase = qh + (size_t)bh * SS * DQn;
  const half_t* kbase = kh + (size_t)b * SS * DKn;
  const half_t* vbase = vh + (size_t)b * SS * DVn;
  const float scale = 0.08838834764831845f;  // 128^-0.5

  // Q fragments resident in VGPRs (16 rows x 128 d = 4 A-frags)
  v16h qf[4];
  {
    const half_t* qrow = qbase + (size_t)(s0 + (w << 4) + ln) * DQn;
    for (int c = 0; c < 4; ++c) {
      v8h c0 = *(const v8h*)(qrow + 32 * c + (hf << 3));
      v8h c1 = *(const v8h*)(qrow + 32 * c + 16 + (hf << 3));
      for (int e = 0; e < 8; ++e) { qf[c][e] = c0[e]; qf[c][e + 8] = c1[e]; }
    }
  }

  v8f zero = {0.f, 0.f, 0.f, 0.f, 0.f, 0.f, 0.f, 0.f};
  v8f yacc[12];
  for (int i = 0; i < 12; ++i) yacc[i] = zero;
  float mrun[8], lrun[8];
  for (int r = 0; r < 8; ++r) { mrun[r] = -1e30f; lrun[r] = 0.f; }

  for (int jt = 0; jt < SS / 32; ++jt) {
    const int j0 = jt * 32;
    // stage V tile (32 keys x 192) transposed into LDS
    for (int i = 0; i < 6; ++i) {
      int u = t + i * 128;
      int kr = u / 24, cg = (u % 24) * 8;
      v8h val = *(const v8h*)(vbase + (size_t)(j0 + kr) * DVn + cg);
      for (int j = 0; j < 8; ++j) vtl[cg + j][kr] = val[j];
    }
    __syncthreads();

    // S = Q K^T  (8 WMMA per wave)
    v8f sacc[2];
    for (int js = 0; js < 2; ++js) {
      sacc[js] = zero;
      const half_t* krow = kbase + (size_t)(j0 + (js << 4) + ln) * DKn;
      for (int c = 0; c < 4; ++c) {
        v16h kf;  // B-frag: lane = key col, elem e -> d = 32c + (e%8) + 16*(e/8) + 8*hf
        v8h c0 = *(const v8h*)(krow + 32 * c + (hf << 3));
        v8h c1 = *(const v8h*)(krow + 32 * c + 16 + (hf << 3));
        for (int e = 0; e < 8; ++e) { kf[e] = c0[e]; kf[e + 8] = c1[e]; }
        sacc[js] = __builtin_amdgcn_wmma_f32_16x16x32_f16(
            false, qf[c], false, kf, (short)0, sacc[js], false, false);
      }
    }

    // bias + softcap + online softmax (row reductions across 16 lanes)
    float p0[8], p1[8], alpha[8];
    for (int r = 0; r < 8; ++r) {
      int sr = s0 + (w << 4) + r + (hf << 3);
      float l0 = sacc[0][r] * scale + bias[(size_t)sr * SS + j0 + ln];
      float l1 = sacc[1][r] * scale + bias[(size_t)sr * SS + j0 + 16 + ln];
      l0 = tanhf(l0 * (1.f / SOFTCAP_F)) * SOFTCAP_F;
      l1 = tanhf(l1 * (1.f / SOFTCAP_F)) * SOFTCAP_F;
      float mx = fmaxf(l0, l1);
      for (int m = 8; m > 0; m >>= 1) mx = fmaxf(mx, __shfl_xor(mx, m, 32));
      float mnew = fmaxf(mrun[r], mx);
      float e0 = expf(l0 - mnew), e1 = expf(l1 - mnew);
      float ls = e0 + e1;
      for (int m = 8; m > 0; m >>= 1) ls += __shfl_xor(ls, m, 32);
      alpha[r] = expf(mrun[r] - mnew);
      lrun[r] = lrun[r] * alpha[r] + ls;
      mrun[r] = mnew;
      p0[r] = e0; p1[r] = e1;
    }
    for (int i = 0; i < 12; ++i)
      for (int r = 0; r < 8; ++r) yacc[i][r] *= alpha[r];

    // round-trip P through per-wave LDS to re-layout C->A fragment
    for (int r = 0; r < 8; ++r) {
      int mrow = r + (hf << 3);
      pbuf[w][mrow][ln]      = (half_t)p0[r];
      pbuf[w][mrow][16 + ln] = (half_t)p1[r];
    }
    __syncthreads();
    v16h pf;
    {
      v8h c0 = *(const v8h*)&pbuf[w][ln][hf << 3];
      v8h c1 = *(const v8h*)&pbuf[w][ln][16 + (hf << 3)];
      for (int e = 0; e < 8; ++e) { pf[e] = c0[e]; pf[e + 8] = c1[e]; }
    }
    // Y += P V  (12 WMMA per wave)
    for (int vt = 0; vt < 12; ++vt) {
      int c = (vt << 4) + ln;
      v16h vf;
      v8h c0 = *(const v8h*)&vtl[c][hf << 3];
      v8h c1 = *(const v8h*)&vtl[c][16 + (hf << 3)];
      for (int e = 0; e < 8; ++e) { vf[e] = c0[e]; vf[e + 8] = c1[e]; }
      yacc[vt] = __builtin_amdgcn_wmma_f32_16x16x32_f16(
          false, pf, false, vf, (short)0, yacc[vt], false, false);
    }
    __syncthreads();
  }

  // normalize and write y in (B,S,H*DV) f16 layout for the out-projection
  for (int vt = 0; vt < 12; ++vt) {
    int c = (vt << 4) + ln;
    for (int r = 0; r < 8; ++r) {
      int sr = s0 + (w << 4) + r + (hf << 3);
      float v = yacc[vt][r] / lrun[r];
      yh[((size_t)(b * SS + sr)) * (HH * DVn) + h * DVn + c] = (half_t)v;
    }
  }
}

// ---------------------------------------------------------------------------
extern "C" void kernel_launch(void* const* d_in, const int* in_sizes, int n_in,
                              void* d_out, int out_size, void* d_ws, size_t ws_size,
                              hipStream_t stream) {
  (void)in_sizes; (void)n_in; (void)out_size; (void)ws_size;
  const float* x   = (const float*)d_in[0];
  const float* ab  = (const float*)d_in[1];
  const float* g1  = (const float*)d_in[2];
  const float* b1  = (const float*)d_in[3];
  const float* rr1 = (const float*)d_in[4];
  const float* Wq  = (const float*)d_in[5];
  const float* Wk  = (const float*)d_in[6];
  const float* Wv  = (const float*)d_in[7];
  const float* qg  = (const float*)d_in[8];
  const float* qb  = (const float*)d_in[9];
  const float* kg  = (const float*)d_in[10];
  const float* kb  = (const float*)d_in[11];
  const float* vg  = (const float*)d_in[12];
  const float* vb  = (const float*)d_in[13];
  const float* Wo  = (const float*)d_in[14];
  const float* bo  = (const float*)d_in[15];
  const float* g2  = (const float*)d_in[16];
  const float* b2  = (const float*)d_in[17];
  const float* rr2 = (const float*)d_in[18];
  float* out = (float*)d_out;

  char* ws = (char*)d_ws;
  size_t off = 0;
  half_t* xn   = (half_t*)(ws + off); off += (size_t)MM * DD * 2;       // 12.6 MB
  half_t* wqkv = (half_t*)(ws + off); off += (size_t)DD * NQKV * 2;     //  4.1 MB
  float*  qkv  = (float*) (ws + off); off += (size_t)MM * NQKV * 4;     // 22.0 MB
  half_t* qh   = (half_t*)(ws + off); off += (size_t)BB * HH * SS * DQn * 2; // 8.4 MB
  half_t* kh   = (half_t*)(ws + off); off += (size_t)BB * SS * DKn * 2; //  1.0 MB
  half_t* vh   = (half_t*)(ws + off); off += (size_t)BB * SS * DVn * 2; //  1.6 MB
  half_t* yh   = (half_t*)(ws + off); off += (size_t)MM * DD * 2;       // 12.6 MB
  half_t* woh  = (half_t*)(ws + off); off += (size_t)DD * DD * 2;       //  4.7 MB

  prenorm_cast<<<(MM * DD + 255) / 256, 256, 0, stream>>>(x, g1, b1, rr1, xn, MM * DD, DD);
  pack_qkv<<<(DD * NQKV + 255) / 256, 256, 0, stream>>>(Wq, Wk, Wv, wqkv, DD * NQKV);
  cast_f32_f16<<<(DD * DD + 255) / 256, 256, 0, stream>>>(Wo, woh, DD * DD);

  dim3 gq(MM / 64, NQKV / 64);   // 64 x 21
  gemm_f16<<<gq, 128, 0, stream>>>(xn, wqkv, qkv, MM, NQKV, DD,
                                   nullptr, nullptr, nullptr, nullptr);

  q_post<<<(BB * HH * SS) / 4, 128, 0, stream>>>(qkv, qg, qb, qh);
  k_post<<<MM / 4, 128, 0, stream>>>(qkv, kg, kb, kh);
  v_post<<<MM / 4, 128, 0, stream>>>(qkv, vg, vb, vh);

  attention<<<dim3(SS / 64, BB * HH), 128, 0, stream>>>(qh, kh, vh, ab, yh);

  dim3 go(MM / 64, DD / 64);     // 64 x 24
  gemm_f16<<<go, 128, 0, stream>>>(yh, woh, out, MM, DD, DD, bo, g2, b2, rr2);
}
